// QMOIMultimodalFusion_18794776888076
// MI455X (gfx1250) — matrix operand order repaired and checked by reading, not compiled, not run on hardware
//
#include <hip/hip_runtime.h>
#include <hip/hip_bf16.h>
#include <cstdint>

// ---------------------------------------------------------------------------
// QMOIMultimodalFusion for MI455X (gfx1250, wave32, WMMA).
// Only token 0 of each modality's attention output reaches d_out, so we prune:
//   - full GEMMs (bf16 WMMA): 3x modality embed (TDM tensor_load_to_lds path),
//     3x K + 3x V (global_load_async_to_lds_b128 double-buffered path)
//   - tiny row-0 work: q0/Wo projections, single-query attention, LN, MLP head
// ---------------------------------------------------------------------------

typedef __bf16   v8bf  __attribute__((ext_vector_type(8)));
typedef __bf16   v16bf __attribute__((ext_vector_type(16)));
typedef float    v8f   __attribute__((ext_vector_type(8)));
typedef uint32_t u32x4 __attribute__((ext_vector_type(4)));
typedef uint32_t u32x8 __attribute__((ext_vector_type(8)));

// ---------------------------------------------------------------------------
// CDNA5 async / TDM primitives
// ---------------------------------------------------------------------------
__device__ __forceinline__ void async_load_b128(uint32_t lds_addr,
                                                const __bf16* gaddr)
{
    // GLOBAL_LOAD_ASYNC_TO_LDS_B128, GV mode: vdst = LDS byte address,
    // vaddr = 64-bit global address. Tracked by ASYNCcnt.
    asm volatile("global_load_async_to_lds_b128 %0, %1, off"
                 :: "v"(lds_addr), "v"(gaddr)
                 : "memory");
}

__device__ __forceinline__ void wait_asynccnt0()
{
    asm volatile("s_wait_asynccnt 0x0" ::: "memory");
}

// Build D# (groups 0+1) per CDNA5 ISA and issue
// TENSOR_LOAD_TO_LDS for a 128 x 32 bf16 tile with LDS row padding
// 32 elems (64B = 16 DWORDs) + 16 elems pad (8 DWORDs) => 48-elem rows.
__device__ __forceinline__ void tdm_load_tile(uint32_t lds_addr,
                                              const __bf16* gptr,
                                              uint32_t rowStrideElems)
{
    const uint64_t ga = (uint64_t)(uintptr_t)gptr;
    u32x4 g0;
    g0.x = 1u;                                             // count=1, user mode
    g0.y = lds_addr;                                       // lds_addr[31:0]
    g0.z = (uint32_t)ga;                                   // global_addr[31:0]
    g0.w = (uint32_t)((ga >> 32) & 0x01FFFFFFu)            // global_addr[56:32]
         | (2u << 30);                                     // type = 2 (image)
    u32x8 g1;
    g1.s0 = (1u << 16)                                     // data_size = 2B
          | (1u << 20)                                     // pad_enable
          | (3u << 22)                                     // pad_interval: 16 DW
          | (7u << 25);                                    // pad_amount: 8 DW
    g1.s1 = (32u << 16);                                   // tensor_dim0 = 32
    g1.s2 = (128u << 16);                                  // tensor_dim1 = 128
    g1.s3 = (32u << 16);                                   // tile_dim0 = 32
    g1.s4 = 128u;                                          // tile_dim1 = 128
    g1.s5 = rowStrideElems;                                // dim0_stride lo32
    g1.s6 = 0u;
    g1.s7 = 0u;
    asm volatile("tensor_load_to_lds %0, %1"
                 :: "s"(g0), "s"(g1)
                 : "memory");
}

// ---------------------------------------------------------------------------
// f32 -> bf16 cast (vectorized x4)
// ---------------------------------------------------------------------------
__global__ __launch_bounds__(256)
void k_cast_bf16(const float* __restrict__ src, __bf16* __restrict__ dst, long n)
{
    long i = ((long)blockIdx.x * blockDim.x + threadIdx.x) * 4;
    if (i + 3 < n) {
        float4 v = *(const float4*)(src + i);
        dst[i + 0] = (__bf16)v.x;
        dst[i + 1] = (__bf16)v.y;
        dst[i + 2] = (__bf16)v.z;
        dst[i + 3] = (__bf16)v.w;
    }
}

// ---------------------------------------------------------------------------
// W (K x N, f32, row-major) -> Wt (N x K, bf16).  32x32 LDS tile transpose.
// ---------------------------------------------------------------------------
__global__ __launch_bounds__(256)
void k_transpose_cast(const float* __restrict__ W, __bf16* __restrict__ Wt,
                      int Kd, int N)
{
    __shared__ float t[32][33];
    const int n0 = blockIdx.x * 32;
    const int k0 = blockIdx.y * 32;
    const int tx = threadIdx.x & 31;
    const int ty = threadIdx.x >> 5;            // 0..7
#pragma unroll
    for (int i = ty; i < 32; i += 8)
        t[i][tx] = W[(long)(k0 + i) * N + (n0 + tx)];
    __syncthreads();
#pragma unroll
    for (int i = ty; i < 32; i += 8)
        Wt[(long)(n0 + i) * Kd + (k0 + tx)] = (__bf16)t[tx][i];
}

// ---------------------------------------------------------------------------
// Shared GEMM compute stage: fragments from LDS (documented gfx1250 16-bit
// A/B lane layouts) + 2x4 v_wmma_f32_16x16x32_bf16, then bias epilogue.
//   A 16x32: lanes 0-15 row M=lane, K=0..7 & 16..23 ; lanes 16-31 K=8..15 & 24..31
//   B 32x16: lanes 0-15 col N=lane, K=0..7 & 16..23 ; lanes 16-31 K=8..15 & 24..31
// ---------------------------------------------------------------------------
__device__ __forceinline__ void gemm_tile_mma(const __bf16 (*As)[48],
                                              const __bf16 (*Bs)[48],
                                              int rA0, int rB0, int kA, int kB,
                                              v8f acc[2][4])
{
    v16bf af[2], bfv[4];
#pragma unroll
    for (int tm = 0; tm < 2; ++tm) {
        const __bf16* p = &As[rA0 + tm * 16][kA];
        v8bf lo = *(const v8bf*)p;
        v8bf hi = *(const v8bf*)(p + 16);
        af[tm] = __builtin_shufflevector(lo, hi, 0,1,2,3,4,5,6,7,
                                         8,9,10,11,12,13,14,15);
    }
#pragma unroll
    for (int tn = 0; tn < 4; ++tn) {
        const __bf16* p = &Bs[rB0 + tn * 16][kB];
        v8bf lo = *(const v8bf*)p;
        v8bf hi = *(const v8bf*)(p + 16);
        bfv[tn] = __builtin_shufflevector(lo, hi, 0,1,2,3,4,5,6,7,
                                          8,9,10,11,12,13,14,15);
    }
#pragma unroll
    for (int tm = 0; tm < 2; ++tm)
#pragma unroll
        for (int tn = 0; tn < 4; ++tn)
            acc[tm][tn] = __builtin_amdgcn_wmma_f32_16x16x32_bf16(
                false, af[tm], false, bfv[tn],
                (short)0, acc[tm][tn], false, false);
}

__device__ __forceinline__ void gemm_epilogue(const float* __restrict__ bias,
                                              __bf16* __restrict__ C,
                                              long bm, long bn, int wm, int wn,
                                              int lane, int N, v8f acc[2][4])
{
    const int rowLane = (lane >> 4) * 8;
    const int colLane = lane & 15;
#pragma unroll
    for (int tm = 0; tm < 2; ++tm)
#pragma unroll
        for (int tn = 0; tn < 4; ++tn) {
            const long col = bn + wn * 64 + tn * 16 + colLane;
            const float bv = bias[col];
#pragma unroll
            for (int r = 0; r < 8; ++r) {
                const long row = bm + wm * 32 + tm * 16 + rowLane + r;
                C[row * (long)N + col] = (__bf16)(acc[tm][tn][r] + bv);
            }
        }
}

// ---------------------------------------------------------------------------
// GEMM variant 1: async-to-LDS double-buffered pipeline.
// C(bf16, MxN) = A(bf16, MxK) @ Bt(bf16, NxK)^T + bias. Tile 128x128, K-step 32.
// Next K-tile streams into the alternate LDS buffer via ASYNCcnt DMA while the
// current buffer feeds the WMMAs.
// ---------------------------------------------------------------------------
__global__ __launch_bounds__(256)
void k_gemm_bf16_async(const __bf16* __restrict__ A, const __bf16* __restrict__ Bt,
                       const float* __restrict__ bias, __bf16* __restrict__ C,
                       int M, int N, int K)
{
    __shared__ __align__(16) __bf16 As[2][128][48];
    __shared__ __align__(16) __bf16 Bs[2][128][48];

    const int tid  = threadIdx.x;
    const int lane = tid & 31;
    const int wid  = tid >> 5;
    const int wm   = wid & 3;
    const int wn   = wid >> 2;
    const long bm  = (long)blockIdx.x * 128;
    const long bn  = (long)blockIdx.y * 128;

    v8f acc[2][4];
#pragma unroll
    for (int i = 0; i < 2; ++i)
#pragma unroll
        for (int j = 0; j < 4; ++j)
            acc[i][j] = v8f{0.f, 0.f, 0.f, 0.f, 0.f, 0.f, 0.f, 0.f};

    // per-thread 16-byte chunks: row = tid>>1, 16 cols starting at (tid&1)*16
    const int lr = tid >> 1;
    const int lc = (tid & 1) << 4;
    const __bf16* gA = A  + (bm + lr) * K + lc;
    const __bf16* gB = Bt + (bn + lr) * K + lc;
    const uint32_t ldsA[2] = { (uint32_t)(uintptr_t)&As[0][lr][lc],
                               (uint32_t)(uintptr_t)&As[1][lr][lc] };
    const uint32_t ldsB[2] = { (uint32_t)(uintptr_t)&Bs[0][lr][lc],
                               (uint32_t)(uintptr_t)&Bs[1][lr][lc] };

    const int kA  = (lane & 16) ? 8  : 0;
    const int kB  = (lane & 16) ? 8  : 0;
    const int rA0 = wm * 32 + (lane & 15);
    const int rB0 = wn * 64 + (lane & 15);

    // prologue: DMA tile 0 into buffer 0
    async_load_b128(ldsA[0],      gA);
    async_load_b128(ldsA[0] + 16, gA + 8);
    async_load_b128(ldsB[0],      gB);
    async_load_b128(ldsB[0] + 16, gB + 8);
    wait_asynccnt0();
    __syncthreads();

    for (int k0 = 0; k0 < K; k0 += 32) {
        const int cur = (k0 >> 5) & 1;
        const int nxt = cur ^ 1;
        if (k0 + 32 < K) {                 // prefetch next tile via async DMA
            async_load_b128(ldsA[nxt],      gA + k0 + 32);
            async_load_b128(ldsA[nxt] + 16, gA + k0 + 40);
            async_load_b128(ldsB[nxt],      gB + k0 + 32);
            async_load_b128(ldsB[nxt] + 16, gB + k0 + 40);
        }
        gemm_tile_mma(As[cur], Bs[cur], rA0, rB0, kA, kB, acc);
        wait_asynccnt0();                  // my wave's DMAs into nxt are done
        __syncthreads();                   // everyone done reading cur / writing nxt
    }

    gemm_epilogue(bias, C, bm, bn, wm, wn, lane, N, acc);
}

// ---------------------------------------------------------------------------
// GEMM variant 2: TDM descriptor DMA (tensor_load_to_lds). Wave 0 issues one
// descriptor per 128x32 tile (hardware applies the 48-elem LDS row padding),
// waits with s_wait_tensorcnt, other waves sync at the workgroup barrier.
// ---------------------------------------------------------------------------
__global__ __launch_bounds__(256)
void k_gemm_bf16_tdm(const __bf16* __restrict__ A, const __bf16* __restrict__ Bt,
                     const float* __restrict__ bias, __bf16* __restrict__ C,
                     int M, int N, int K)
{
    __shared__ __align__(16) __bf16 As[2][128][48];
    __shared__ __align__(16) __bf16 Bs[2][128][48];

    const int tid  = threadIdx.x;
    const int lane = tid & 31;
    const int wid  = tid >> 5;
    const int wm   = wid & 3;
    const int wn   = wid >> 2;
    const long bm  = (long)blockIdx.x * 128;
    const long bn  = (long)blockIdx.y * 128;

    v8f acc[2][4];
#pragma unroll
    for (int i = 0; i < 2; ++i)
#pragma unroll
        for (int j = 0; j < 4; ++j)
            acc[i][j] = v8f{0.f, 0.f, 0.f, 0.f, 0.f, 0.f, 0.f, 0.f};

    const __bf16* gA0 = A  + bm * K;     // tile (row 0, k 0)
    const __bf16* gB0 = Bt + bn * K;
    const uint32_t ldsA[2] = { (uint32_t)(uintptr_t)&As[0][0][0],
                               (uint32_t)(uintptr_t)&As[1][0][0] };
    const uint32_t ldsB[2] = { (uint32_t)(uintptr_t)&Bs[0][0][0],
                               (uint32_t)(uintptr_t)&Bs[1][0][0] };

    const int kA  = (lane & 16) ? 8  : 0;
    const int kB  = (lane & 16) ? 8  : 0;
    const int rA0 = wm * 32 + (lane & 15);
    const int rB0 = wn * 64 + (lane & 15);

    if (tid < 32) {                        // wave 0 drives the TDM
        tdm_load_tile(ldsA[0], gA0, (uint32_t)K);
        tdm_load_tile(ldsB[0], gB0, (uint32_t)K);
        __builtin_amdgcn_s_wait_tensorcnt(0);
    }
    __syncthreads();

    for (int k0 = 0; k0 < K; k0 += 32) {
        const int cur = (k0 >> 5) & 1;
        const int nxt = cur ^ 1;
        if ((k0 + 32 < K) && tid < 32) {   // TDM-prefetch next K tile
            tdm_load_tile(ldsA[nxt], gA0 + k0 + 32, (uint32_t)K);
            tdm_load_tile(ldsB[nxt], gB0 + k0 + 32, (uint32_t)K);
        }
        gemm_tile_mma(As[cur], Bs[cur], rA0, rB0, kA, kB, acc);
        if (tid < 32) __builtin_amdgcn_s_wait_tensorcnt(0);
        __syncthreads();
    }

    gemm_epilogue(bias, C, bm, bn, wm, wn, lane, N, acc);
}

// ---------------------------------------------------------------------------
// Small column GEMM: Y[r,c] = act( sum_k X[r,k]*W[k,c] + bias[c] )
// Used for q0, Wo(row 0), and the 4-row fused MLP. W reads are coalesced.
// ---------------------------------------------------------------------------
template <typename TI>
__global__ __launch_bounds__(256)
void k_colgemm(const TI* __restrict__ X, long rowStride,
               const float* __restrict__ W, const float* __restrict__ bias,
               float* __restrict__ Y, int Kd, int N, int act)
{
    const int r = blockIdx.y;
    const int c = blockIdx.x * blockDim.x + threadIdx.x;
    const TI* x = X + (long)r * rowStride;
    float acc = 0.f;
    for (int k = 0; k < Kd; ++k)
        acc = fmaf((float)x[k], W[(long)k * N + c], acc);
    acc += bias[c];
    if (act)  // exact GELU: x * 0.5 * (1 + erf(x/sqrt(2)))
        acc = 0.5f * acc * (1.f + erff(acc * 0.70710678f));
    Y[(long)r * N + c] = acc;
}

// ---------------------------------------------------------------------------
// Single-query attention (query = token 0; RoPE at pos 0 is identity for q).
// One block per (b,h). scores[j] = (q0 . rope_j(k_j)) / 8 + q0 . rel[j+S-1]
// Softmax over j, out = sum_j p_j * v_j.
// ---------------------------------------------------------------------------
__global__ __launch_bounds__(256)
void k_attn_row0(const float* __restrict__ q0, const __bf16* __restrict__ Kb,
                 const __bf16* __restrict__ Vb, const float* __restrict__ rel,
                 float* __restrict__ o)
{
    const int b   = blockIdx.x >> 4;
    const int h   = blockIdx.x & 15;
    const int tid = threadIdx.x;
    __shared__ float qs[64];
    __shared__ float red[256];
    __shared__ float osh[64];

    if (tid < 64) { qs[tid] = q0[b * 1024 + h * 64 + tid]; osh[tid] = 0.f; }
    __syncthreads();

    float sc[4];
    float m = -3.4e38f;
#pragma unroll
    for (int jj = 0; jj < 4; ++jj) {
        const int j = jj * 256 + tid;
        const __bf16* kp = Kb + ((long)(b * 1024 + j)) * 1024 + h * 64;
        const float*  rp = rel + (long)(j + 1023) * 64;
        float dqk = 0.f, drl = 0.f;
#pragma unroll
        for (int d = 0; d < 32; ++d) {
            const float k1 = (float)kp[d];
            const float k2 = (float)kp[d + 32];
            // inv_freq = 10000^(-d/32) = exp(-d * ln(10000)/32)
            const float ang = (float)j * __expf(-0.28782313f * (float)d);
            float sn, cs;
            __sincosf(ang, &sn, &cs);
            dqk += qs[d]      * (k1 * cs - k2 * sn);
            dqk += qs[d + 32] * (k1 * sn + k2 * cs);
            drl += qs[d] * rp[d] + qs[d + 32] * rp[d + 32];
        }
        sc[jj] = 0.125f * dqk + drl;   // scale = D^-0.5 = 1/8 (rel term unscaled)
        m = fmaxf(m, sc[jj]);
    }
    red[tid] = m;
    __syncthreads();
    for (int s = 128; s > 0; s >>= 1) {
        if (tid < s) red[tid] = fmaxf(red[tid], red[tid + s]);
        __syncthreads();
    }
    const float mx = red[0];
    __syncthreads();

    float p[4];
    float psum = 0.f;
#pragma unroll
    for (int jj = 0; jj < 4; ++jj) { p[jj] = __expf(sc[jj] - mx); psum += p[jj]; }
    red[tid] = psum;
    __syncthreads();
    for (int s = 128; s > 0; s >>= 1) {
        if (tid < s) red[tid] += red[tid + s];
        __syncthreads();
    }
    const float denom = red[0];

    float pv[64];
#pragma unroll
    for (int d = 0; d < 64; ++d) pv[d] = 0.f;
#pragma unroll
    for (int jj = 0; jj < 4; ++jj) {
        const int j = jj * 256 + tid;
        const __bf16* vp = Vb + ((long)(b * 1024 + j)) * 1024 + h * 64;
#pragma unroll
        for (int d = 0; d < 64; ++d) pv[d] += p[jj] * (float)vp[d];
    }
#pragma unroll
    for (int d = 0; d < 64; ++d) atomicAdd(&osh[d], pv[d]);   // ds_add_f32
    __syncthreads();
    if (tid < 64) o[b * 1024 + h * 64 + tid] = osh[tid] / denom;
}

// ---------------------------------------------------------------------------
// LayerNorm one row of 1024, write into fused slot.
// ---------------------------------------------------------------------------
__global__ __launch_bounds__(256)
void k_ln_row(const float* __restrict__ y, const float* __restrict__ g,
              const float* __restrict__ bn, float* __restrict__ fused, int slot)
{
    const int b   = blockIdx.x;
    const int tid = threadIdx.x;
    __shared__ float red[256];
    const float* yr = y + b * 1024;
    float s = 0.f;
    for (int i = tid; i < 1024; i += 256) s += yr[i];
    red[tid] = s;
    __syncthreads();
    for (int st = 128; st > 0; st >>= 1) {
        if (tid < st) red[tid] += red[tid + st];
        __syncthreads();
    }
    const float mu = red[0] * (1.f / 1024.f);
    __syncthreads();
    float v = 0.f;
    for (int i = tid; i < 1024; i += 256) { float d = yr[i] - mu; v += d * d; }
    red[tid] = v;
    __syncthreads();
    for (int st = 128; st > 0; st >>= 1) {
        if (tid < st) red[tid] += red[tid + st];
        __syncthreads();
    }
    const float inv = rsqrtf(red[0] * (1.f / 1024.f) + 1e-5f);
    for (int i = tid; i < 1024; i += 256)
        fused[(long)b * 3072 + slot * 1024 + i] = (yr[i] - mu) * inv * g[i] + bn[i];
}

// ---------------------------------------------------------------------------
extern "C" void kernel_launch(void* const* d_in, const int* in_sizes, int n_in,
                              void* d_out, int out_size, void* d_ws, size_t ws_size,
                              hipStream_t stream)
{
    const int  B = 4, S = 1024, E = 1024, H = 16;
    const long BSE = (long)B * S * E;        // 4,194,304
    const long EE  = (long)E * E;

    auto f32 = [&](int i) { return (const float*)d_in[i]; };

    // bump allocator over d_ws
    char*  ws  = (char*)d_ws;
    size_t off = 0;
    auto alloc = [&](size_t bytes) -> void* {
        void* p = ws + off;
        off = (off + bytes + 255) & ~(size_t)255;
        return p;
    };

    __bf16* xb[3];  for (int m = 0; m < 3; ++m) xb[m]  = (__bf16*)alloc(BSE * 2);
    __bf16* wT[5];  for (int w = 0; w < 5; ++w) wT[w]  = (__bf16*)alloc(EE * 2);
    __bf16* emb[3]; for (int m = 0; m < 3; ++m) emb[m] = (__bf16*)alloc(BSE * 2);
    __bf16* Kb[3];  for (int m = 0; m < 3; ++m) Kb[m]  = (__bf16*)alloc(BSE * 2);
    __bf16* Vb[3];  for (int m = 0; m < 3; ++m) Vb[m]  = (__bf16*)alloc(BSE * 2);
    float*  q0[3];  for (int m = 0; m < 3; ++m) q0[m]  = (float*)alloc((long)B * E * 4);
    float*  at0[3]; for (int m = 0; m < 3; ++m) at0[m] = (float*)alloc((long)B * E * 4);
    float*  yv[3];  for (int m = 0; m < 3; ++m) yv[m]  = (float*)alloc((long)B * E * 4);
    float*  fused = (float*)alloc((long)B * 3 * E * 4);
    float*  hb    = (float*)alloc((long)B * E * 4);

    const dim3 blk(256);

    // 1) cast modality inputs to bf16
    const int castGrid = (int)(BSE / (256 * 4));
    for (int m = 0; m < 3; ++m)
        k_cast_bf16<<<castGrid, blk, 0, stream>>>(f32(m), xb[m], BSE);

    // 2) transpose+cast weights to bf16 NxK: Wt, Wi, Wa, Wk, Wv
    const int widx[5] = {3, 5, 7, 11, 13};
    const dim3 tgrid(E / 32, E / 32);
    for (int w = 0; w < 5; ++w)
        k_transpose_cast<<<tgrid, blk, 0, stream>>>(f32(widx[w]), wT[w], E, E);

    // 3) WMMA GEMMs: embed via TDM descriptor DMA, K/V via async-to-LDS DMA
    const dim3 ggrid((B * S) / 128, E / 128);
    const int  bidx[3] = {4, 6, 8};   // bt, bi, ba
    for (int m = 0; m < 3; ++m)
        k_gemm_bf16_tdm<<<ggrid, blk, 0, stream>>>(xb[m], wT[m], f32(bidx[m]),
                                                   emb[m], B * S, E, E);
    for (int m = 0; m < 3; ++m) {
        k_gemm_bf16_async<<<ggrid, blk, 0, stream>>>(emb[m], wT[3], f32(12),
                                                     Kb[m], B * S, E, E);
        k_gemm_bf16_async<<<ggrid, blk, 0, stream>>>(emb[m], wT[4], f32(14),
                                                     Vb[m], B * S, E, E);
    }

    // 4) q0 projection (token 0 only), attention row 0, Wo, LN -> fused
    const dim3 cgrid(E / 256, B);
    for (int m = 0; m < 3; ++m) {
        k_colgemm<__bf16><<<cgrid, blk, 0, stream>>>(emb[m], (long)S * E,
                                                     f32(9), f32(10), q0[m], E, E, 0);
        k_attn_row0<<<B * H, blk, 0, stream>>>(q0[m], Kb[m], Vb[m], f32(17), at0[m]);
        k_colgemm<float><<<cgrid, blk, 0, stream>>>(at0[m], (long)E,
                                                    f32(15), f32(16), yv[m], E, E, 0);
        k_ln_row<<<B, blk, 0, stream>>>(yv[m], f32(18 + 2 * m), f32(19 + 2 * m),
                                        fused, m);
    }

    // 5) fused MLP head: gelu(fused @ Wf1 + bf1) @ Wf2 + bf2 -> d_out (4x1024 f32)
    k_colgemm<float><<<cgrid, blk, 0, stream>>>(fused, 3L * E, f32(24), f32(25),
                                                hb, 3 * E, E, 1);
    k_colgemm<float><<<cgrid, blk, 0, stream>>>(hb, (long)E, f32(26), f32(27),
                                                (float*)d_out, E, E, 0);
}
